// QWEN_MLP_77111842832762
// MI455X (gfx1250) — matrix operand order repaired and checked by reading, not compiled, not run on hardware
//
#include <hip/hip_runtime.h>

#define TT 16384   // tokens
#define HH 768     // hidden
#define II 2048    // intermediate
#define KTOP 8

typedef __attribute__((ext_vector_type(16))) __bf16 v16bf;
typedef __attribute__((ext_vector_type(8)))  __bf16 v8bf;
typedef __attribute__((ext_vector_type(8)))  float  v8f;

#if defined(__gfx1250__) &&                                        \
    __has_builtin(__builtin_amdgcn_global_load_async_to_lds_b128) && \
    __has_builtin(__builtin_amdgcn_s_wait_asynccnt)
#define USE_ASYNC 1
#else
#define USE_ASYNC 0
#endif

union FragU { v16bf v; v8bf h[2]; };

__device__ __forceinline__ __bf16 f2bf(float f) {
  unsigned u = __builtin_bit_cast(unsigned, f);
  u = (u + 0x7FFFu + ((u >> 16) & 1u)) >> 16;
  unsigned short s = (unsigned short)u;
  return __builtin_bit_cast(__bf16, s);
}

#if USE_ASYNC
// Builtin signature (from hipcc diagnostic): param0 = int __vector(4) * AS(1),
// param1 (by symmetry) = int __vector(4) * AS(3), then imm offset, imm cpol.
typedef int v4i32 __attribute__((vector_size(16)));
typedef __attribute__((address_space(1))) v4i32* as1_v4i;
typedef __attribute__((address_space(3))) v4i32* as3_v4i;

// 16-byte global -> LDS async copy (ASYNCcnt-tracked, per ISA 08_async_tensor §4)
__device__ __forceinline__ void asy16(const __bf16* g, __bf16* l) {
  __builtin_amdgcn_global_load_async_to_lds_b128((as1_v4i)g, (as3_v4i)l, 0, 0);
}
#endif

__global__ void qwen_cvt_bf16(const float* __restrict__ src,
                              __bf16* __restrict__ dst, int n) {
  int stride = gridDim.x * blockDim.x;
  for (int i = blockIdx.x * blockDim.x + threadIdx.x; i < n; i += stride)
    dst[i] = f2bf(src[i]);
}

// src [K][N] f32  ->  dst [N][K] bf16 (coalesced both sides via LDS tile)
__global__ __launch_bounds__(256) void qwen_cvt_tr(const float* __restrict__ src,
                                                   __bf16* __restrict__ dst,
                                                   int K, int N) {
  __shared__ __bf16 t[32][33];
  const int kb = blockIdx.x * 32, nb = blockIdx.y * 32;
  const int tx = threadIdx.x & 31, ty = threadIdx.x >> 5;  // 32 x 8
#pragma unroll
  for (int i = 0; i < 32; i += 8)
    t[ty + i][tx] = f2bf(src[(size_t)(kb + ty + i) * N + nb + tx]);
  __syncthreads();
#pragma unroll
  for (int i = 0; i < 32; i += 8)
    dst[(size_t)(nb + ty + i) * K + kb + tx] = t[tx][ty + i];
}

__global__ void qwen_combine(const int* __restrict__ mask,
                             const float* __restrict__ rw,
                             float* __restrict__ comb) {
  int t = blockIdx.x * blockDim.x + threadIdx.x;
  if (t >= TT) return;
  float s = 0.f;
#pragma unroll
  for (int j = 0; j < KTOP; ++j)
    s += mask[j * TT + t] ? rw[t * KTOP + j] : 0.f;
  comb[t] = s;
}

// A fragment (16x32 bf16): lane L holds row fr; K chunks at kb and 16+kb, kb=(L/16)*8.
__device__ __forceinline__ v16bf load_fragA(const __bf16* tile, int row, int lane) {
  const int kb = (lane >> 4) << 3;
  const __bf16* p = tile + row * 32;
  FragU f;
  f.h[0] = *(const v8bf*)(p + kb);
  f.h[1] = *(const v8bf*)(p + 16 + kb);
  return f.v;
}

// B fragment (32x16 bf16): lane L holds col L%16; K chunks at kb and 16+kb, kb=(L/16)*8.
// Tile stored [N][K] (K-contiguous) thanks to pre-transposed weights.
__device__ __forceinline__ v16bf load_fragB(const __bf16* tile, int col, int lane) {
  const __bf16* p = tile + col * 32 + ((lane >> 4) << 3);
  FragU f;
  f.h[0] = *(const v8bf*)p;
  f.h[1] = *(const v8bf*)(p + 16);
  return f.v;
}

// GEMM1: h[T,I] = silu(X@Wg) * (X@Wu). Block tile: 128 tokens x 64 cols.
// WgT/WuT are pre-transposed [I][H] so both A and B tiles are straight copies.
__global__ __launch_bounds__(256, 2) void qwen_gemm1(
    const __bf16* __restrict__ X, const __bf16* __restrict__ WgT,
    const __bf16* __restrict__ WuT, __bf16* __restrict__ Hout) {
  __shared__ __align__(16) __bf16 sX[2][128 * 32];
  __shared__ __align__(16) __bf16 sWg[2][64 * 32];
  __shared__ __align__(16) __bf16 sWu[2][64 * 32];

  const int tid = threadIdx.x;
  const int lane = tid & 31;
  const int wid = tid >> 5;
  const int waveM = wid & 3;   // 4 waves along M
  const int waveN = wid >> 2;  // 2 waves along N
  const int rowBase = blockIdx.x * 128;
  const int colBase = blockIdx.y * 64;
  const int fr = lane & 15;

  v8f accG[2][2], accU[2][2];
#pragma unroll
  for (int mi = 0; mi < 2; ++mi)
#pragma unroll
    for (int ni = 0; ni < 2; ++ni) {
      accG[mi][ni] = {0.f,0.f,0.f,0.f,0.f,0.f,0.f,0.f};
      accU[mi][ni] = {0.f,0.f,0.f,0.f,0.f,0.f,0.f,0.f};
    }

  // 16B-chunk copy geometry (8 bf16 per chunk, 4 chunks per 32-wide row)
  const int xr0 = tid >> 2;              // X rows 0..63
  const int xr1 = xr0 + 64;              // X rows 64..127
  const int xc  = (tid & 3) << 3;
  const int wn  = tid >> 2;              // weight tile row (N) 0..63
  const int wk  = (tid & 3) << 3;

  auto compute = [&](int buf) {
    v16bf a[2], bg[2], bu[2];
#pragma unroll
    for (int mi = 0; mi < 2; ++mi)
      a[mi] = load_fragA(sX[buf], waveM * 32 + mi * 16 + fr, lane);
#pragma unroll
    for (int ni = 0; ni < 2; ++ni) {
      bg[ni] = load_fragB(sWg[buf], waveN * 32 + ni * 16 + fr, lane);
      bu[ni] = load_fragB(sWu[buf], waveN * 32 + ni * 16 + fr, lane);
    }
#pragma unroll
    for (int mi = 0; mi < 2; ++mi)
#pragma unroll
      for (int ni = 0; ni < 2; ++ni) {
        accG[mi][ni] = __builtin_amdgcn_wmma_f32_16x16x32_bf16(
            false, a[mi], false, bg[ni], (short)0, accG[mi][ni], false, false);
        accU[mi][ni] = __builtin_amdgcn_wmma_f32_16x16x32_bf16(
            false, a[mi], false, bu[ni], (short)0, accU[mi][ni], false, false);
      }
  };

#if USE_ASYNC
  auto stage = [&](int k0, int buf) {
    asy16(X   + (size_t)(rowBase + xr0) * HH + k0 + xc, &sX[buf][xr0 * 32 + xc]);
    asy16(X   + (size_t)(rowBase + xr1) * HH + k0 + xc, &sX[buf][xr1 * 32 + xc]);
    asy16(WgT + (size_t)(colBase + wn) * HH + k0 + wk, &sWg[buf][wn * 32 + wk]);
    asy16(WuT + (size_t)(colBase + wn) * HH + k0 + wk, &sWu[buf][wn * 32 + wk]);
  };
  stage(0, 0);
  int buf = 0;
  for (int k0 = 0; k0 < HH; k0 += 32, buf ^= 1) {
    if (k0 + 32 < HH) {
      stage(k0 + 32, buf ^ 1);                 // prefetch next tile into other buffer
      if (k0 + 64 < HH)
        __builtin_prefetch(X + (size_t)(rowBase + xr0) * HH + k0 + 64 + xc, 0, 0);
      __builtin_amdgcn_s_wait_asynccnt(4);     // oldest 4 (current tile) done
    } else {
      __builtin_amdgcn_s_wait_asynccnt(0);
    }
    __syncthreads();
    compute(buf);
    __syncthreads();
  }
#else
  for (int k0 = 0; k0 < HH; k0 += 32) {
    uint4 x0 = *(const uint4*)(X   + (size_t)(rowBase + xr0) * HH + k0 + xc);
    uint4 x1 = *(const uint4*)(X   + (size_t)(rowBase + xr1) * HH + k0 + xc);
    uint4 g0 = *(const uint4*)(WgT + (size_t)(colBase + wn) * HH + k0 + wk);
    uint4 u0 = *(const uint4*)(WuT + (size_t)(colBase + wn) * HH + k0 + wk);
    __syncthreads();
    *(uint4*)&sX[0][xr0 * 32 + xc] = x0;
    *(uint4*)&sX[0][xr1 * 32 + xc] = x1;
    *(uint4*)&sWg[0][wn * 32 + wk] = g0;
    *(uint4*)&sWu[0][wn * 32 + wk] = u0;
    __syncthreads();
    compute(0);
  }
#endif

  // C/D layout: lane L -> col = L%16; VGPR v -> row = v + 8*(L/16)
  const int rh = (lane >> 4) << 3;
#pragma unroll
  for (int mi = 0; mi < 2; ++mi)
#pragma unroll
    for (int ni = 0; ni < 2; ++ni) {
      int col = colBase + waveN * 32 + ni * 16 + fr;
#pragma unroll
      for (int v = 0; v < 8; ++v) {
        int row = rowBase + waveM * 32 + mi * 16 + rh + v;
        float g = accG[mi][ni][v];
        float u = accU[mi][ni][v];
        float sig = __builtin_amdgcn_rcpf(1.f + __expf(-g));  // v_rcp_f32, no IEEE div
        Hout[(size_t)row * II + col] = f2bf(g * sig * u);
      }
    }
}

// GEMM2: out[T,H] = final + (h @ Wd) * combine[t]. WdT pre-transposed [H][I].
__global__ __launch_bounds__(256, 2) void qwen_gemm2(
    const __bf16* __restrict__ Hb, const __bf16* __restrict__ WdT,
    const float* __restrict__ comb, const float* __restrict__ fin,
    float* __restrict__ Out) {
  __shared__ __align__(16) __bf16 sA[2][128 * 32];
  __shared__ __align__(16) __bf16 sB[2][64 * 32];
  __shared__ float sC[128];

  const int tid = threadIdx.x;
  const int lane = tid & 31;
  const int wid = tid >> 5;
  const int waveM = wid & 3;
  const int waveN = wid >> 2;
  const int rowBase = blockIdx.x * 128;
  const int colBase = blockIdx.y * 64;
  const int fr = lane & 15;

  if (tid < 128) sC[tid] = comb[rowBase + tid];

  v8f acc[2][2];
#pragma unroll
  for (int mi = 0; mi < 2; ++mi)
#pragma unroll
    for (int ni = 0; ni < 2; ++ni)
      acc[mi][ni] = {0.f,0.f,0.f,0.f,0.f,0.f,0.f,0.f};

  const int xr0 = tid >> 2;
  const int xr1 = xr0 + 64;
  const int xc  = (tid & 3) << 3;
  const int wn  = tid >> 2;
  const int wk  = (tid & 3) << 3;

  auto compute = [&](int buf) {
    v16bf a[2], b[2];
#pragma unroll
    for (int mi = 0; mi < 2; ++mi)
      a[mi] = load_fragA(sA[buf], waveM * 32 + mi * 16 + fr, lane);
#pragma unroll
    for (int ni = 0; ni < 2; ++ni)
      b[ni] = load_fragB(sB[buf], waveN * 32 + ni * 16 + fr, lane);
#pragma unroll
    for (int mi = 0; mi < 2; ++mi)
#pragma unroll
      for (int ni = 0; ni < 2; ++ni)
        acc[mi][ni] = __builtin_amdgcn_wmma_f32_16x16x32_bf16(
            false, a[mi], false, b[ni], (short)0, acc[mi][ni], false, false);
  };

#if USE_ASYNC
  auto stage = [&](int k0, int buf) {
    asy16(Hb  + (size_t)(rowBase + xr0) * II + k0 + xc, &sA[buf][xr0 * 32 + xc]);
    asy16(Hb  + (size_t)(rowBase + xr1) * II + k0 + xc, &sA[buf][xr1 * 32 + xc]);
    asy16(WdT + (size_t)(colBase + wn) * II + k0 + wk, &sB[buf][wn * 32 + wk]);
  };
  stage(0, 0);
  int buf = 0;
  for (int k0 = 0; k0 < II; k0 += 32, buf ^= 1) {
    if (k0 + 32 < II) {
      stage(k0 + 32, buf ^ 1);
      if (k0 + 64 < II)
        __builtin_prefetch(Hb + (size_t)(rowBase + xr0) * II + k0 + 64 + xc, 0, 0);
      __builtin_amdgcn_s_wait_asynccnt(3);
    } else {
      __builtin_amdgcn_s_wait_asynccnt(0);
    }
    __syncthreads();
    compute(buf);
    __syncthreads();
  }
#else
  for (int k0 = 0; k0 < II; k0 += 32) {
    uint4 a0 = *(const uint4*)(Hb  + (size_t)(rowBase + xr0) * II + k0 + xc);
    uint4 a1 = *(const uint4*)(Hb  + (size_t)(rowBase + xr1) * II + k0 + xc);
    uint4 b0 = *(const uint4*)(WdT + (size_t)(colBase + wn) * II + k0 + wk);
    __syncthreads();
    *(uint4*)&sA[0][xr0 * 32 + xc] = a0;
    *(uint4*)&sA[0][xr1 * 32 + xc] = a1;
    *(uint4*)&sB[0][wn * 32 + wk] = b0;
    __syncthreads();
    compute(0);
  }
#endif

  const int rh = (lane >> 4) << 3;
#pragma unroll
  for (int mi = 0; mi < 2; ++mi)
#pragma unroll
    for (int ni = 0; ni < 2; ++ni) {
      int col = colBase + waveN * 32 + ni * 16 + fr;
#pragma unroll
      for (int v = 0; v < 8; ++v) {
        int rloc = waveM * 32 + mi * 16 + rh + v;
        size_t idx = (size_t)(rowBase + rloc) * HH + col;
        Out[idx] = fin[idx] + acc[mi][ni][v] * sC[rloc];
      }
    }
}

extern "C" void kernel_launch(void* const* d_in, const int* in_sizes, int n_in,
                              void* d_out, int out_size, void* d_ws, size_t ws_size,
                              hipStream_t stream) {
  (void)in_sizes; (void)n_in; (void)out_size; (void)ws_size;
  const float* hs  = (const float*)d_in[0];   // hidden_states   [T,H] f32
  const float* rw  = (const float*)d_in[1];   // routing_weights [T,8] f32
  const float* fin = (const float*)d_in[2];   // final_hidden    [T,H] f32
  const float* gw  = (const float*)d_in[3];   // gate_w [H,I] f32
  const float* uw  = (const float*)d_in[4];   // up_w   [H,I] f32
  const float* dw  = (const float*)d_in[5];   // down_w [I,H] f32
  const int*   mk  = (const int*)d_in[6];     // expert_mask [8,T] i32
  float* out = (float*)d_out;

  // workspace layout (bytes)
  char* w = (char*)d_ws;
  __bf16* Xb  = (__bf16*)(w);                 // [T][H]  25165824 B
  __bf16* WgT = (__bf16*)(w + 25165824);      // [I][H]   3145728 B
  __bf16* WuT = (__bf16*)(w + 28311552);      // [I][H]
  __bf16* WdT = (__bf16*)(w + 31457280);      // [H][I]
  __bf16* Hb  = (__bf16*)(w + 34603008);      // [T][I]  67108864 B
  float*  cb  = (float*) (w + 101711872);     // [T] f32

  qwen_cvt_bf16<<<2048, 256, 0, stream>>>(hs, Xb, TT * HH);
  qwen_cvt_tr<<<dim3(HH / 32, II / 32), 256, 0, stream>>>(gw, WgT, HH, II);
  qwen_cvt_tr<<<dim3(HH / 32, II / 32), 256, 0, stream>>>(uw, WuT, HH, II);
  qwen_cvt_tr<<<dim3(II / 32, HH / 32), 256, 0, stream>>>(dw, WdT, II, HH);
  qwen_combine<<<TT / 256, 256, 0, stream>>>(mk, rw, cb);
  qwen_gemm1<<<dim3(TT / 128, II / 64), 256, 0, stream>>>(Xb, WgT, WuT, Hb);
  qwen_gemm2<<<dim3(TT / 128, HH / 64), 256, 0, stream>>>(Hb, WdT, cb, fin, out);
}
